// QuantumTSTransformer_16054587753011
// MI455X (gfx1250) — matrix operand order repaired and checked by reading, not compiled, not run on hardware
//
#include <hip/hip_runtime.h>
#include <hip/hip_bf16.h>
#include <math.h>

#define NQ        12
#define DIM       4096          // 2^12
#define TSTEPS    32
#define BATCH     64
#define NROTS     96
#define QFF_NROTS 48
#define BT        (BATCH * TSTEPS)   // 2048
#define ROT_SCALE 6.2831853071795864769f

typedef __attribute__((ext_vector_type(16))) _Float16 v16h;
typedef __attribute__((ext_vector_type(8)))  float    v8f;

// ---------------------------------------------------------------------------
// Kernel 1: params = sigmoid(x @ W_proj^T + b_proj) * 2pi   via WMMA f16->f32
// M = 2048 (b*T), K = 64 (two k-blocks of 32), N = 96.
// 768 16x16 tiles -> 96 blocks x 8 waves, one tile per wave.
// ---------------------------------------------------------------------------
__global__ __launch_bounds__(256)
void proj_kernel(const float* __restrict__ x,      // [2048, 64]
                 const float* __restrict__ Wp,     // [96, 64]
                 const float* __restrict__ bp,     // [96]
                 float* __restrict__ params) {     // [2048, 96]
    const int wid  = threadIdx.x >> 5;
    const int lane = threadIdx.x & 31;
    const int tile = blockIdx.x * 8 + wid;   // 0..767
    const int tm   = tile / 6;               // 0..127
    const int tn   = tile % 6;               // 0..5

    const int mrow = tm * 16 + (lane & 15);
    const int ncol = tn * 16 + (lane & 15);
    const int aKlo = (lane < 16) ? 0 : 8;    // A 16x32 f16 layout (ISA 7.12.2)
    const int bKlo = (lane < 16) ? 0 : 8;    // B 32x16 f16 layout

    const float* xr = x  + mrow * 64;
    const float* wr = Wp + ncol * 64;

    v16h a0, a1, b0, b1;
#pragma unroll
    for (int h = 0; h < 16; ++h) {
        const int ka = aKlo + h + ((h >= 8) ? 8 : 0);   // K in [0,32)
        a0[h] = (_Float16)xr[ka];
        a1[h] = (_Float16)xr[32 + ka];
        const int kb = bKlo + h + ((h >= 8) ? 8 : 0);   // K in [0,32)
        b0[h] = (_Float16)wr[kb];                       // B[k][n] = W_proj[n][k]
        b1[h] = (_Float16)wr[32 + kb];
    }

    v8f c = {};
    c = __builtin_amdgcn_wmma_f32_16x16x32_f16(false, a0, false, b0, (short)0, c, false, false);
    c = __builtin_amdgcn_wmma_f32_16x16x32_f16(false, a1, false, b1, (short)0, c, false, false);

    const float bias = bp[ncol];
#pragma unroll
    for (int v = 0; v < 8; ++v) {
        const int m = tm * 16 + v + ((lane < 16) ? 0 : 8);   // C/D layout
        const float z  = c[v] + bias;
        const float sg = 1.0f / (1.0f + expf(-z));
        params[m * NROTS + ncol] = sg * ROT_SCALE;
    }
}

// ---------------------------------------------------------------------------
// LDS-resident quantum gate helpers (wire 0 = most significant bit)
// ---------------------------------------------------------------------------
__device__ __forceinline__ void ry_gate(float* re, float* im, int wire,
                                        float c, float s, int tid) {
    const int sw = 1 << (NQ - 1 - wire);
#pragma unroll
    for (int p = tid; p < (DIM / 2); p += 256) {
        const int i0 = ((p & ~(sw - 1)) << 1) | (p & (sw - 1));
        const int i1 = i0 + sw;
        const float r0 = re[i0], m0 = im[i0];
        const float r1 = re[i1], m1 = im[i1];
        re[i0] = c * r0 - s * r1;  im[i0] = c * m0 - s * m1;
        re[i1] = s * r0 + c * r1;  im[i1] = s * m0 + c * m1;
    }
}

__device__ __forceinline__ void crx_gate(float* re, float* im, int ctrl, int tgt,
                                         float c, float s, int tid) {
    const int sc = 1 << (NQ - 1 - ctrl);
    const int st = 1 << (NQ - 1 - tgt);
    const int hi = (sc > st) ? sc : st;
    const int lo = (sc > st) ? st : sc;
#pragma unroll
    for (int p = tid; p < (DIM / 4); p += 256) {
        int i = ((p & ~(lo - 1)) << 1) | (p & (lo - 1));
        i     = ((i & ~(hi - 1)) << 1) | (i & (hi - 1));
        const int i0 = i | sc;          // control=1, target=0
        const int i1 = i0 | st;         // control=1, target=1
        const float r0 = re[i0], m0 = im[i0];
        const float r1 = re[i1], m1 = im[i1];
        // a0' = c*a0 - i*s*a1 ; a1' = c*a1 - i*s*a0
        re[i0] = c * r0 + s * m1;  im[i0] = c * m0 - s * r1;
        re[i1] = c * r1 + s * m0;  im[i1] = c * m1 - s * r0;
    }
}

// sim14 ansatz: RY ring, reversed CRX ring, RY ring, rotated CRX ring
__device__ void sim14_lds(float* re, float* im,
                          const float* cs, const float* sn,
                          int layers, int tid) {
    int idx = 0;
    for (int L = 0; L < layers; ++L) {
        for (int i = 0; i < NQ; ++i) {
            ry_gate(re, im, i, cs[idx], sn[idx], tid); ++idx; __syncthreads();
        }
        for (int i = NQ - 1; i >= 0; --i) {
            crx_gate(re, im, i, (i + 1) % NQ, cs[idx], sn[idx], tid); ++idx; __syncthreads();
        }
        for (int i = 0; i < NQ; ++i) {
            ry_gate(re, im, i, cs[idx], sn[idx], tid); ++idx; __syncthreads();
        }
        // order: [n-1, 0, 1, ..., n-2], target = (i-1) mod n
        crx_gate(re, im, NQ - 1, NQ - 2, cs[idx], sn[idx], tid); ++idx; __syncthreads();
        for (int i = 0; i < NQ - 1; ++i) {
            crx_gate(re, im, i, (i + NQ - 1) % NQ, cs[idx], sn[idx], tid); ++idx; __syncthreads();
        }
    }
}

// ---------------------------------------------------------------------------
// Kernel 0: work = |0..0>,  acc = poly[0] * |0..0>
// ---------------------------------------------------------------------------
__global__ __launch_bounds__(256)
void init_kernel(const float* __restrict__ poly,
                 float* __restrict__ work, float* __restrict__ acc) {
    const int b = blockIdx.x;
    const int j = blockIdx.y * 256 + threadIdx.x;        // 0..4095
    const size_t o = (size_t)b * (2 * DIM) + 2 * j;
    work[o]     = (j == 0) ? 1.0f : 0.0f;
    work[o + 1] = 0.0f;
    acc[o]      = (j == 0) ? poly[0] : 0.0f;
    acc[o + 1]  = 0.0f;
}

// ---------------------------------------------------------------------------
// Kernel 2: for each (b,t): ev[b,t,:] = sim14_2layer(work[b,:], params[b,t,:])
// One workgroup per state; state resident in 32KB LDS.
// ---------------------------------------------------------------------------
__global__ __launch_bounds__(256)
void sim_step_kernel(const float* __restrict__ work,    // [64, 4096] cplx
                     const float* __restrict__ params,  // [2048, 96]
                     float* __restrict__ ev) {          // [2048, 4096] cplx
    __shared__ float re[DIM];
    __shared__ float im[DIM];
    __shared__ float cs[NROTS];
    __shared__ float sn[NROTS];

    const int bt  = blockIdx.x;         // 0..2047
    const int b   = bt >> 5;            // / TSTEPS
    const int tid = threadIdx.x;

    const float* src = work + (size_t)b * (2 * DIM);
    for (int j = tid; j < DIM; j += 256) {
        re[j] = src[2 * j];
        im[j] = src[2 * j + 1];
    }
    if (tid < NROTS) {
        const float h = params[bt * NROTS + tid] * 0.5f;
        cs[tid] = cosf(h);
        sn[tid] = sinf(h);
    }
    __syncthreads();

    sim14_lds(re, im, cs, sn, 2, tid);

    float* dst = ev + (size_t)bt * (2 * DIM);
    for (int j = tid; j < DIM; j += 256) {
        dst[2 * j]     = re[j];
        dst[2 * j + 1] = im[j];
    }
}

// ---------------------------------------------------------------------------
// Kernel 3: work[b,:] = sum_t mix[t] * ev[b,t,:] ;  acc += poly[k] * work
// Fixed-order reduction -> deterministic (no atomics).
// ---------------------------------------------------------------------------
__global__ __launch_bounds__(256)
void mix_reduce_kernel(const float* __restrict__ ev,
                       const float* __restrict__ mix_re,
                       const float* __restrict__ mix_im,
                       const float* __restrict__ poly, int k,
                       float* __restrict__ work, float* __restrict__ acc) {
    const int b = blockIdx.x;
    const int j = blockIdx.y * 256 + threadIdx.x;        // 0..4095
    const float* base = ev + (size_t)b * TSTEPS * (2 * DIM) + 2 * j;
    float pr = 0.0f, pi = 0.0f;
    for (int t = 0; t < TSTEPS; ++t) {
        const float er = base[(size_t)t * (2 * DIM)];
        const float ei = base[(size_t)t * (2 * DIM) + 1];
        const float mr = mix_re[t], mi = mix_im[t];
        pr += mr * er - mi * ei;
        pi += mr * ei + mi * er;
    }
    const size_t o = (size_t)b * (2 * DIM) + 2 * j;
    work[o]     = pr;
    work[o + 1] = pi;
    const float pk = poly[k];
    acc[o]     += pk * pr;
    acc[o + 1] += pk * pi;
}

// ---------------------------------------------------------------------------
// block-wide sum (fixed order, deterministic)
// ---------------------------------------------------------------------------
__device__ float block_reduce_sum(float v, float* red, int tid) {
    __syncthreads();            // protect red[] from prior use
    red[tid] = v;
    __syncthreads();
    for (int s = 128; s > 0; s >>= 1) {
        if (tid < s) red[tid] += red[tid + s];
        __syncthreads();
    }
    return red[0];
}

// ---------------------------------------------------------------------------
// Kernel 4: normalize acc, apply 1-layer sim14(qff), Pauli expvals, out GEMM
// ---------------------------------------------------------------------------
__global__ __launch_bounds__(256)
void finalize_kernel(const float* __restrict__ acc,
                     const float* __restrict__ poly,
                     const float* __restrict__ qff,
                     const float* __restrict__ W_out,   // [8, 36]
                     const float* __restrict__ b_out,   // [8]
                     float* __restrict__ out) {         // [64, 8]
    __shared__ float re[DIM];
    __shared__ float im[DIM];
    __shared__ float cs[QFF_NROTS];
    __shared__ float sn[QFF_NROTS];
    __shared__ float red[256];
    __shared__ float exps[3 * NQ];

    const int b   = blockIdx.x;
    const int tid = threadIdx.x;

    const float sumabs = fabsf(poly[0]) + fabsf(poly[1]) + fabsf(poly[2]) +
                         fabsf(poly[3]) + fabsf(poly[4]);
    const float inv = 1.0f / sumabs;

    const float* src = acc + (size_t)b * (2 * DIM);
    float local = 0.0f;
    for (int j = tid; j < DIM; j += 256) {
        const float r = src[2 * j] * inv;
        const float m = src[2 * j + 1] * inv;
        re[j] = r; im[j] = m;
        local += r * r + m * m;
    }
    const float nrm = sqrtf(block_reduce_sum(local, red, tid));
    const float sc  = 1.0f / (nrm + 1e-9f);
    for (int j = tid; j < DIM; j += 256) { re[j] *= sc; im[j] *= sc; }
    if (tid < QFF_NROTS) {
        const float h = qff[tid] * 0.5f;
        cs[tid] = cosf(h);
        sn[tid] = sinf(h);
    }
    __syncthreads();

    sim14_lds(re, im, cs, sn, 1, tid);   // ends with __syncthreads()

    // Pauli expectation values: X0..X11, Y0..Y11, Z0..Z11
    for (int w = 0; w < NQ; ++w) {
        const int sw = 1 << (NQ - 1 - w);
        float xr = 0.0f, xi = 0.0f, zz = 0.0f;
        for (int p = tid; p < (DIM / 2); p += 256) {
            const int i0 = ((p & ~(sw - 1)) << 1) | (p & (sw - 1));
            const int i1 = i0 + sw;
            const float r0 = re[i0], m0 = im[i0];
            const float r1 = re[i1], m1 = im[i1];
            xr += r0 * r1 + m0 * m1;     // Re(conj(a0)*a1)
            xi += r0 * m1 - m0 * r1;     // Im(conj(a0)*a1)
            zz += r0 * r0 + m0 * m0 - r1 * r1 - m1 * m1;
        }
        float t;
        t = block_reduce_sum(xr, red, tid); if (tid == 0) exps[w]          = 2.0f * t;
        t = block_reduce_sum(xi, red, tid); if (tid == 0) exps[NQ + w]     = 2.0f * t;
        t = block_reduce_sum(zz, red, tid); if (tid == 0) exps[2 * NQ + w] = t;
    }
    __syncthreads();

    if (tid < 8) {
        float o = b_out[tid];
        for (int j = 0; j < 3 * NQ; ++j) o += exps[j] * W_out[tid * (3 * NQ) + j];
        out[b * 8 + tid] = o;
    }
}

// ---------------------------------------------------------------------------
extern "C" void kernel_launch(void* const* d_in, const int* in_sizes, int n_in,
                              void* d_out, int out_size, void* d_ws, size_t ws_size,
                              hipStream_t stream) {
    const float* x      = (const float*)d_in[0];   // [64, 32, 64]
    const float* W_proj = (const float*)d_in[1];   // [96, 64]
    const float* b_proj = (const float*)d_in[2];   // [96]
    const float* poly   = (const float*)d_in[3];   // [5]
    const float* mix_re = (const float*)d_in[4];   // [32]
    const float* mix_im = (const float*)d_in[5];   // [32]
    const float* qff    = (const float*)d_in[6];   // [48]
    const float* W_out  = (const float*)d_in[7];   // [8, 36]
    const float* b_out  = (const float*)d_in[8];   // [8]
    float* out = (float*)d_out;                    // [64, 8]

    // Workspace layout (floats):
    //   params : 2048*96            =   196,608
    //   work   : 64*4096*2          =   524,288
    //   acc    : 64*4096*2          =   524,288
    //   ev     : 2048*4096*2        = 16,777,216   (~64 MB; fits in L2)
    float* params = (float*)d_ws;
    float* work   = params + (size_t)BT * NROTS;
    float* acc    = work + (size_t)BATCH * 2 * DIM;
    float* ev     = acc  + (size_t)BATCH * 2 * DIM;

    proj_kernel<<<96, 256, 0, stream>>>(x, W_proj, b_proj, params);
    init_kernel<<<dim3(BATCH, DIM / 256), 256, 0, stream>>>(poly, work, acc);

    for (int k = 1; k <= 4; ++k) {
        sim_step_kernel<<<BT, 256, 0, stream>>>(work, params, ev);
        mix_reduce_kernel<<<dim3(BATCH, DIM / 256), 256, 0, stream>>>(
            ev, mix_re, mix_im, poly, k, work, acc);
    }

    finalize_kernel<<<BATCH, 256, 0, stream>>>(acc, poly, qff, W_out, b_out, out);
}
